// SequenceAtomTransformerBlock_59940563583357
// MI455X (gfx1250) — matrix operand order repaired and checked by reading, not compiled, not run on hardware
//
#include <hip/hip_runtime.h>
#include <cstdint>
#include <cstddef>

#define B_   4
#define NA_  8192
#define NR_  512
#define NB_  256       // NA/WQ
#define WQ_  32
#define WK_  128
#define NH_  4
#define CH_  32

typedef __bf16 bf16;
typedef bf16  v16bf __attribute__((ext_vector_type(16)));
typedef bf16  v8bf  __attribute__((ext_vector_type(8)));
typedef bf16  v4bf  __attribute__((ext_vector_type(4)));
typedef float v8f   __attribute__((ext_vector_type(8)));

__device__ __forceinline__ float sigf(float x) { return 1.f / (1.f + __expf(-x)); }

__device__ __forceinline__ v8f wmma_bf16(v16bf a, v16bf b, v8f c) {
  return __builtin_amdgcn_wmma_f32_16x16x32_bf16(false, a, false, b, (short)0, c, false, false);
}

// ---- CDNA5 async global->LDS copy (16B per lane), ASYNCcnt tracked ----
__device__ __forceinline__ void async_copy_b128(uint32_t lds_off, const void* gptr) {
  asm volatile("global_load_async_to_lds_b128 %0, %1, off"
               :: "v"(lds_off), "v"((uint64_t)(uintptr_t)gptr) : "memory");
}
__device__ __forceinline__ void wait_async0() {
  asm volatile("s_wait_asynccnt 0x0" ::: "memory");
}

// ---- WMMA fragment loaders (contiguous k = 16*(lane/16) + e in both A and B) ----
// A 16x32 bf16, source row-major [m][k]: per lane = two contiguous 16B chunks.
__device__ __forceinline__ v16bf load_A(const bf16* src, int ld) {
  int lane = threadIdx.x & 31;
  int m = lane & 15;
  int khi = (lane >> 4) << 4;
  const bf16* p = src + m * ld + khi;
  v8bf lo = *(const v8bf*)p;
  v8bf hi = *(const v8bf*)(p + 8);
  v16bf f;
#pragma unroll
  for (int e = 0; e < 8; ++e) { f[e] = lo[e]; f[8 + e] = hi[e]; }
  return f;
}
// B fragment where each lane's 16 elements are contiguous 32B in memory.
__device__ __forceinline__ v16bf load_frag32B(const bf16* p) {
  return *(const v16bf*)p;
}
// B fragment from pre-swizzled weights: one contiguous 32B vector per lane.
__device__ __forceinline__ v16bf load_B_sw(const bf16* tile_base) {
  int lane = threadIdx.x & 31;
  return *(const v16bf*)(tile_base + lane * 16);
}

// =====================================================================
// Kernel W: swizzle f32 weight (KxN) into B-fragment-order bf16.
// =====================================================================
__global__ __launch_bounds__(256) void k_wswizzle(const float* __restrict__ W, int K, int N,
                                                  bf16* __restrict__ out)
{
  int gid = blockIdx.x * 256 + threadIdx.x;     // < K*N
  int e = gid & 15;
  int lane = (gid >> 4) & 31;
  int tile = gid >> 9;
  int kt = tile % (K >> 5);
  int nt = tile / (K >> 5);
  int k = kt * 32 + ((lane >> 4) << 4) + e;
  int n = nt * 16 + (lane & 15);
  out[gid] = (bf16)W[(size_t)k * N + n];
}

// =====================================================================
// Kernel A: per-residue conditioning (LN + six 256->128 matvecs)
// =====================================================================
__device__ __forceinline__ float mv256(const float* src, const float* __restrict__ W, int c) {
  float acc = 0.f;
#pragma unroll 8
  for (int i = 0; i < 256; ++i) acc += src[i] * W[(size_t)i * 128 + c];
  return acc;
}

__global__ __launch_bounds__(256) void k_res_prep(
    const float* __restrict__ res, const float* __restrict__ adaln_ln_w,
    const float* __restrict__ condW, const float* __restrict__ condb,
    const float* __restrict__ condnbW,
    const float* __restrict__ tr_ln_w, const float* __restrict__ trcondW,
    const float* __restrict__ trcondb, const float* __restrict__ trcondnbW,
    const float* __restrict__ sgW, const float* __restrict__ sgb,
    const float* __restrict__ tgW, const float* __restrict__ tgb,
    float* __restrict__ sig_gate, float* __restrict__ bias_o,
    float* __restrict__ sig_g2, float* __restrict__ b2_o,
    float* __restrict__ sig_sg, float* __restrict__ sig_tg)
{
  __shared__ float sh_row[256], sh_c1[256], sh_c2[256], red[9];
  int t = threadIdx.x;
  int r = blockIdx.x;
  const float* row = res + (size_t)r * 256;
  float v = row[t];
  sh_row[t] = v;
  float s = v;
#pragma unroll
  for (int o = 16; o > 0; o >>= 1) s += __shfl_xor(s, o, 32);
  if ((t & 31) == 0) red[t >> 5] = s;
  __syncthreads();
  if (t == 0) { float a = 0; for (int i = 0; i < 8; ++i) a += red[i]; red[8] = a * (1.f / 256.f); }
  __syncthreads();
  float mean = red[8];
  float d = v - mean, vv = d * d;
#pragma unroll
  for (int o = 16; o > 0; o >>= 1) vv += __shfl_xor(vv, o, 32);
  __syncthreads();
  if ((t & 31) == 0) red[t >> 5] = vv;
  __syncthreads();
  if (t == 0) { float a = 0; for (int i = 0; i < 8; ++i) a += red[i]; red[8] = rsqrtf(a * (1.f / 256.f) + 1e-5f); }
  __syncthreads();
  float rs = red[8];
  float y = d * rs;
  sh_c1[t] = y * adaln_ln_w[t];
  sh_c2[t] = y * tr_ln_w[t];
  __syncthreads();

  int c = t & 127;
  size_t ro = (size_t)r * 128;
  if (t < 128) {
    sig_gate[ro + c] = sigf(mv256(sh_c1, condW, c) + condb[c]);
    sig_g2[ro + c]   = sigf(mv256(sh_c2, trcondW, c) + trcondb[c]);
    sig_sg[ro + c]   = sigf(mv256(sh_row, sgW, c) + sgb[c]);
  } else {
    bias_o[ro + c] = mv256(sh_c1, condnbW, c);
    b2_o[ro + c]   = mv256(sh_c2, trcondnbW, c);
    sig_tg[ro + c] = sigf(mv256(sh_row, tgW, c) + tgb[c]);
  }
}

// =====================================================================
// Kernel A2: n_valid per batch
// =====================================================================
__global__ __launch_bounds__(256) void k_nvalid(const unsigned char* __restrict__ maskp,
                                                int* __restrict__ nv)
{
  __shared__ int red[8];
  int b = blockIdx.x, t = threadIdx.x;
  int s = 0;
  for (int i = t; i < NA_; i += 256) s += maskp[(size_t)b * NA_ + i] ? 1 : 0;
#pragma unroll
  for (int o = 16; o > 0; o >>= 1) s += __shfl_xor(s, o, 32);
  if ((t & 31) == 0) red[t >> 5] = s;
  __syncthreads();
  if (t == 0) { int a = 0; for (int i = 0; i < 8; ++i) a += red[i]; nv[b] = a; }
}

// =====================================================================
// Kernel B: atom LN + adaLN  ->  af (bf16)
// =====================================================================
__global__ __launch_bounds__(256) void k_atom_adaln(
    const float* __restrict__ atomf, const int* __restrict__ res_idx,
    const float* __restrict__ sig_gate, const float* __restrict__ bias_w,
    bf16* __restrict__ af)
{
  int w = threadIdx.x >> 5, lane = threadIdx.x & 31;
  size_t atom = (size_t)blockIdx.x * 8 + w;   // flat row in [0, B*NA)
  int b = (int)(atom / NA_);
  const float* row = atomf + atom * 128;
  float4 xx = *(const float4*)(row + lane * 4);
  float x[4] = { xx.x, xx.y, xx.z, xx.w };
  float s = x[0] + x[1] + x[2] + x[3];
#pragma unroll
  for (int o = 16; o > 0; o >>= 1) s += __shfl_xor(s, o, 32);
  float mean = s * (1.f / 128.f);
  float vv = 0.f;
#pragma unroll
  for (int i = 0; i < 4; ++i) { float d = x[i] - mean; vv += d * d; }
#pragma unroll
  for (int o = 16; o > 0; o >>= 1) vv += __shfl_xor(vv, o, 32);
  float rs = rsqrtf(vv * (1.f / 128.f) + 1e-5f);
  int res = res_idx[atom];
  const float4 g4 = *(const float4*)(sig_gate + ((size_t)b * NR_ + res) * 128 + lane * 4);
  const float4 b4 = *(const float4*)(bias_w + ((size_t)b * NR_ + res) * 128 + lane * 4);
  const float g[4] = { g4.x, g4.y, g4.z, g4.w };
  const float bb[4] = { b4.x, b4.y, b4.z, b4.w };
  v4bf o;
#pragma unroll
  for (int i = 0; i < 4; ++i) o[i] = (bf16)((x[i] - mean) * rs * g[i] + bb[i]);
  *(v4bf*)(af + atom * 128 + lane * 4) = o;
}

// =====================================================================
// Kernel C: fused Q/KV projection GEMM  (M=B*NA, K=128, N=384) via WMMA
// q,k row-major [atom][128]; v written TRANSPOSED: v_t[b][ch][atom].
// grid = (M/16, 6), block = 128 (4 waves, one 16x16 tile each)
// =====================================================================
__global__ __launch_bounds__(128) void k_qkv(
    const bf16* __restrict__ af, const bf16* __restrict__ Wq_s,
    const float* __restrict__ bq, const bf16* __restrict__ Wkv_s,
    bf16* __restrict__ qo, bf16* __restrict__ ko, bf16* __restrict__ vt)
{
  __shared__ __align__(16) bf16 aT[16 * 128];
  int t = threadIdx.x;
  size_t m0 = (size_t)blockIdx.x * 16;
  {
    const uint4* src = (const uint4*)(af + m0 * 128);
    uint4* dst = (uint4*)aT;
    for (int i = t; i < 256; i += 128) dst[i] = src[i];
  }
  __syncthreads();

  int w = t >> 5, lane = t & 31;
  int n0 = blockIdx.y * 64 + w * 16;          // 0..368, never straddles 128/256
  int nn = lane & 15, mlo = (lane >> 4) << 3;
  const bf16* WS;
  int col0;
  if (n0 < 128) { WS = Wq_s;  col0 = n0; }
  else          { WS = Wkv_s; col0 = n0 - 128; }
  const bf16* wtile = WS + ((size_t)(col0 >> 4) * 4) * 512;   // 4 k-tiles of 512 bf16 each

  v8f acc = {};
#pragma unroll
  for (int ks = 0; ks < 128; ks += 32) {
    v16bf a = load_A(aT + ks, 128);
    v16bf bfr = load_B_sw(wtile + (ks >> 5) * 512);
    acc = wmma_bf16(a, bfr, acc);
  }

  // wave-uniform region select: no divergent stores
  int region = n0 >> 7;                 // 0=q, 1=k, 2=v
  if (region == 0) {
    int col = n0 + nn;
    float bqv = bq[col];
#pragma unroll
    for (int r2 = 0; r2 < 8; ++r2)
      qo[(m0 + r2 + mlo) * 128 + col] = (bf16)((acc[r2] + bqv) * 0.17677669529663687f); // 1/sqrt(32)
  } else if (region == 1) {
    int col = n0 - 128 + nn;
#pragma unroll
    for (int r2 = 0; r2 < 8; ++r2)
      ko[(m0 + r2 + mlo) * 128 + col] = (bf16)acc[r2];
  } else {
    int colv = n0 - 256 + nn;
    int b = (int)(m0 / NA_);
    int al = (int)(m0 % NA_);
    v8bf ov;
#pragma unroll
    for (int r2 = 0; r2 < 8; ++r2) ov[r2] = (bf16)acc[r2];
    *(v8bf*)(vt + ((size_t)b * 128 + colv) * NA_ + al + mlo) = ov;   // 8 consecutive atoms
  }
}

// =====================================================================
// Kernel D: pair features + windowed attention (one block per (b, window))
// block = 256 threads (8 waves). K/V B-fragments streamed from global (L2);
// only q + p + scores live in LDS (~109 KB).
// =====================================================================
#define D_CQ    0                      // q bf16 [32][128]            8192
#define D_CP    8192                   // p bf16 [4][32][128]        32768
#define D_CSC   (D_CP + 32768)        // scores f32 [4][32][128]    65536 (out aliases first 16KB)
#define D_CRT   (D_CSC + 65536)       // R^T|t f32 [32][12]          1536
#define D_CRQ   (D_CRT + 1536)        // res of q atoms, int[32]      128
#define D_CC    (D_CRQ + 128)         // consts (160 floats)          640
#define D_CPQ   (D_CC + 640)          // pos_q f32 [32][3]            384
#define D_CPK   (D_CPQ + 384)         // pos_k f32 [128][3]          1536
#define D_CRK   (D_CPK + 1536)        // res of key atoms, int[128]   512
#define D_CQM   (D_CRK + 512)         // q mask int[32]               128
#define D_SMEM  (D_CQM + 128)

__global__ __launch_bounds__(256) void k_attn(
    const float* __restrict__ atomf, const float* __restrict__ atompos,
    const float* __restrict__ rigids, const int* __restrict__ res_idx,
    const unsigned char* __restrict__ maskp, const int* __restrict__ nvalid,
    const float* __restrict__ dvecW, const float* __restrict__ distW,
    const float* __restrict__ lnw, const float* __restrict__ lnb,
    const float* __restrict__ bijW,
    const bf16* __restrict__ qg, const bf16* __restrict__ kg, const bf16* __restrict__ vt,
    const float* __restrict__ sig_sg,
    float* __restrict__ xo, float* __restrict__ apf_out)
{
  __shared__ __align__(16) unsigned char smem[D_SMEM];
  bf16*  qs   = (bf16*)(smem + D_CQ);
  bf16*  ps   = (bf16*)(smem + D_CP);
  float* sc   = (float*)(smem + D_CSC);
  float* outb = (float*)(smem + D_CSC);  // alias: out over scores after softmax
  float* RT   = (float*)(smem + D_CRT);
  int*   resq = (int*)(smem + D_CRQ);
  float* cdvec = (float*)(smem + D_CC);
  float* cdist = cdvec + 48;
  float* clnw  = cdist + 16;
  float* clnb  = clnw + 16;
  float* cbij  = clnb + 16;
  float* posq = (float*)(smem + D_CPQ);
  float* posk = (float*)(smem + D_CPK);
  int*   resk = (int*)(smem + D_CRK);
  int*   qm   = (int*)(smem + D_CQM);

  int t = threadIdx.x, w = t >> 5, lane = t & 31;
  int n = blockIdx.x, b = blockIdx.y;
  size_t abase = (size_t)b * NA_;
  size_t aq0 = abase + n * WQ_;
  bool noclamp = (n * WQ_ + WK_ - 1) < NA_;

  // ---- async-stage q ----
  {
    const unsigned char* gq = (const unsigned char*)(qg + aq0 * 128);
    uint32_t lq = (uint32_t)(uintptr_t)(smem + D_CQ);
    for (int i = t; i < 512; i += 256)
      async_copy_b128(lq + i * 16, gq + i * 16);
  }

  // ---- stage constants / geometry ----
  if (t < 160) {
    float v;
    if (t < 48)       v = dvecW[t];
    else if (t < 64)  v = distW[t - 48];
    else if (t < 80)  v = lnw[t - 64];
    else if (t < 96)  v = lnb[t - 80];
    else              v = bijW[t - 96];
    cdvec[t] = v;   // consts are laid out contiguously
  }
  if (t < 32) {
    size_t aq = aq0 + t;
    int res = res_idx[aq];
    resq[t] = res;
    qm[t] = maskp[aq] ? 1 : 0;
    const float* r7 = rigids + ((size_t)b * NR_ + res) * 7;
    float w0 = r7[0], x0 = r7[1], y0 = r7[2], z0 = r7[3];
    float* M = RT + t * 12;               // M[i][j] = R[j][i]
    M[0] = 1.f - 2.f * (y0 * y0 + z0 * z0); M[1] = 2.f * (x0 * y0 + w0 * z0); M[2] = 2.f * (x0 * z0 - w0 * y0);
    M[3] = 2.f * (x0 * y0 - w0 * z0); M[4] = 1.f - 2.f * (x0 * x0 + z0 * z0); M[5] = 2.f * (y0 * z0 + w0 * x0);
    M[6] = 2.f * (x0 * z0 + w0 * y0); M[7] = 2.f * (y0 * z0 - w0 * x0); M[8] = 1.f - 2.f * (x0 * x0 + y0 * y0);
    M[9] = r7[4]; M[10] = r7[5]; M[11] = r7[6];
  }
  if (t < 96) posq[t] = atompos[aq0 * 3 + t];
  for (int i = t; i < 384; i += 256) {
    int kk = i / 3;
    int ak0 = n * WQ_ + kk; if (ak0 > NA_ - 1) ak0 = NA_ - 1;
    posk[i] = atompos[(abase + ak0) * 3 + (i - kk * 3)];
  }
  if (t < 128) {
    int ak0 = n * WQ_ + t; if (ak0 > NA_ - 1) ak0 = NA_ - 1;
    resk[t] = res_idx[abase + ak0];
  }
  wait_async0();
  __syncthreads();

  // ---- pair features: apf out + (LN(apf)@bijW + mask) into score buffer ----
  int nv = nvalid[b];
  for (int pp = t * 16; pp < t * 16 + 16; ++pp) {
    int qi = pp >> 7, kk = pp & 127;
    const float* M = RT + qi * 12;
    float dx = posq[qi * 3 + 0] - posk[kk * 3 + 0] - M[9];
    float dy = posq[qi * 3 + 1] - posk[kk * 3 + 1] - M[10];
    float dz = posq[qi * 3 + 2] - posk[kk * 3 + 2] - M[11];
    float l0 = M[0] * dx + M[1] * dy + M[2] * dz;
    float l1 = M[3] * dx + M[4] * dy + M[5] * dz;
    float l2 = M[6] * dx + M[7] * dy + M[8] * dz;
    float dist = sqrtf(l0 * l0 + l1 * l1 + l2 * l2);
    float same = (resq[qi] == resk[kk]) ? 1.f : 0.f;
    float av[16]; float msum = 0.f;
#pragma unroll
    for (int c = 0; c < 16; ++c) {
      float vv = (l0 * cdvec[c] + l1 * cdvec[16 + c] + l2 * cdvec[32 + c] + dist * cdist[c]) * same;
      av[c] = vv; msum += vv;
    }
    float* ap = apf_out + ((((size_t)b * NB_ + n) * WQ_ + qi) * WK_ + kk) * 16;
#pragma unroll
    for (int c = 0; c < 16; ++c) ap[c] = av[c];
    float mu = msum * (1.f / 16.f), var = 0.f;
#pragma unroll
    for (int c = 0; c < 16; ++c) { float d = av[c] - mu; var += d * d; }
    float rs = rsqrtf(var * (1.f / 16.f) + 1e-5f);
    float bh0 = 0, bh1 = 0, bh2 = 0, bh3 = 0;
#pragma unroll
    for (int c = 0; c < 16; ++c) {
      float y = (av[c] - mu) * rs * clnw[c] + clnb[c];
      bh0 += y * cbij[c * 4 + 0]; bh1 += y * cbij[c * 4 + 1];
      bh2 += y * cbij[c * 4 + 2]; bh3 += y * cbij[c * 4 + 3];
    }
    float pen = ((n * WQ_ + kk) < nv && qm[qi]) ? 0.f : -100000000.f;
    sc[(0 * 32 + qi) * 128 + kk] = bh0 + pen;
    sc[(1 * 32 + qi) * 128 + kk] = bh1 + pen;
    sc[(2 * 32 + qi) * 128 + kk] = bh2 + pen;
    sc[(3 * 32 + qi) * 128 + kk] = bh3 + pen;
  }
  __syncthreads();

  // ---- QK^T WMMA: wave w covers head w>>1, row-tile w&1, all 8 col-tiles.
  //      B-fragment streamed from global k (contiguous 32B per lane; clamp via row index).
  {
    int h = w >> 1, tm = w & 1;
    int nn = lane & 15, kb0 = (lane >> 4) << 4, mlo = (lane >> 4) << 3;
    v16bf a = load_A(qs + (tm * 16) * 128 + h * CH_, 128);
#pragma unroll
    for (int tn = 0; tn < 8; ++tn) {
      int key = n * WQ_ + tn * 16 + nn; if (key > NA_ - 1) key = NA_ - 1;
      v16bf bfr = load_frag32B(kg + (abase + key) * 128 + h * CH_ + kb0);
      float* scp = sc + (h * 32 + tm * 16) * 128 + tn * 16;
      v8f c0;
#pragma unroll
      for (int r2 = 0; r2 < 8; ++r2) c0[r2] = scp[(r2 + mlo) * 128 + nn];
      v8f d0 = wmma_bf16(a, bfr, c0);
#pragma unroll
      for (int r2 = 0; r2 < 8; ++r2) scp[(r2 + mlo) * 128 + nn] = d0[r2];
    }
  }
  __syncthreads();

  // ---- softmax over k (128), one thread per (h, q) row ----
  if (t < 128) {
    float* row = sc + t * 128;
    float mx = row[0];
    for (int k2 = 1; k2 < 128; ++k2) mx = fmaxf(mx, row[k2]);
    float ssum = 0.f;
    for (int k2 = 0; k2 < 128; ++k2) { float e = __expf(row[k2] - mx); ssum += e; row[k2] = e; }
    float inv = 1.f / ssum;
    bf16* pr = ps + t * 128;
    for (int k2 = 0; k2 < 128; ++k2) pr[k2] = (bf16)(row[k2] * inv);
  }
  __syncthreads();

  // ---- P@V WMMA: 16 tiles (h, tm, tn2) over 8 waves.
  //      B-fragment streamed from transposed global v_t (contiguous 32B per lane).
  {
    int nn = lane & 15, kb0 = (lane >> 4) << 4, mlo = (lane >> 4) << 3;
    for (int tt = w; tt < 16; tt += 8) {
      int h = tt >> 2, tm = (tt >> 1) & 1, tn2 = tt & 1;
      int ch = h * CH_ + tn2 * 16 + nn;
      const bf16* vrow = vt + ((size_t)b * 128 + ch) * NA_;
      v8f acc = {};
#pragma unroll
      for (int ks2 = 0; ks2 < 128; ks2 += 32) {
        v16bf a = load_A(ps + (h * 32 + tm * 16) * 128 + ks2, 128);
        v16bf bfr;
        if (noclamp) {
          bfr = load_frag32B(vrow + n * WQ_ + ks2 + kb0);
        } else {
#pragma unroll
          for (int e = 0; e < 16; ++e) {
            int key = n * WQ_ + ks2 + kb0 + e; if (key > NA_ - 1) key = NA_ - 1;
            bfr[e] = vrow[key];
          }
        }
        acc = wmma_bf16(a, bfr, acc);
      }
      float* op = outb + (tm * 16) * 128 + h * CH_ + tn2 * 16;
#pragma unroll
      for (int r2 = 0; r2 < 8; ++r2) op[(r2 + mlo) * 128 + nn] = acc[r2];
    }
  }
  __syncthreads();

  // ---- epilogue: x = atom_features + out * sigmoid(sgate) * mask ----
  for (int i4 = t; i4 < 1024; i4 += 256) {
    int idx = i4 * 4;
    int qi = idx >> 7, c = idx & 127;
    size_t aq = aq0 + qi;
    int res = resq[qi];
    float4 g4 = *(const float4*)(sig_sg + ((size_t)b * NR_ + res) * 128 + c);
    float4 a4 = *(const float4*)(atomf + aq * 128 + c);
    float mf = qm[qi] ? 1.f : 0.f;
    const float* ob = outb + idx;
    float4 r;
    r.x = a4.x + ob[0] * g4.x * mf;
    r.y = a4.y + ob[1] * g4.y * mf;
    r.z = a4.z + ob[2] * g4.z * mf;
    r.w = a4.w + ob[3] * g4.w * mf;
    *(float4*)(xo + aq * 128 + c) = r;
  }
}

// =====================================================================
// Kernel E: conditioned transition (128 -> 512 -> 128) via WMMA
// block = 256 threads (8 waves), 16 atoms per block
// =====================================================================
__global__ __launch_bounds__(256) void k_trans(
    const bf16* __restrict__ lin1_s, const bf16* __restrict__ lin2_s,
    const bf16* __restrict__ linb_s,
    const float* __restrict__ sig_g2, const float* __restrict__ b2w,
    const float* __restrict__ sig_tg,
    const int* __restrict__ res_idx, const unsigned char* __restrict__ maskp,
    float* __restrict__ xo)
{
  __shared__ __align__(16) bf16 s2[16 * 128];
  __shared__ __align__(16) float hraw[16 * 512];
  __shared__ __align__(16) bf16 hbb[16 * 256];
  __shared__ int resl[16];

  int t = threadIdx.x, w = t >> 5, lane = t & 31;
  size_t m0 = (size_t)blockIdx.x * 16;
  int bb = (int)(m0 / NA_);

  // phase 1: LN(x) * sigmoid(g2) + b2  (2 atoms per wave, 16 lanes each)
  {
    int at = 2 * w + (lane >> 4);
    int l16 = lane & 15;
    size_t arow = m0 + at;
    const float* xr = xo + arow * 128;
    float4 xa = *(const float4*)(xr + l16 * 8);
    float4 xb = *(const float4*)(xr + l16 * 8 + 4);
    float xv[8] = { xa.x, xa.y, xa.z, xa.w, xb.x, xb.y, xb.z, xb.w };
    float s = 0.f;
#pragma unroll
    for (int i = 0; i < 8; ++i) s += xv[i];
#pragma unroll
    for (int o = 8; o > 0; o >>= 1) s += __shfl_xor(s, o, 32);
    float mean = s * (1.f / 128.f);
    float vv = 0.f;
#pragma unroll
    for (int i = 0; i < 8; ++i) { float d = xv[i] - mean; vv += d * d; }
#pragma unroll
    for (int o = 8; o > 0; o >>= 1) vv += __shfl_xor(vv, o, 32);
    float rs = rsqrtf(vv * (1.f / 128.f) + 1e-5f);
    int res = res_idx[arow];
    if (l16 == 0) resl[at] = res;
    const float* g2 = sig_g2 + ((size_t)bb * NR_ + res) * 128;
    const float* b2 = b2w + ((size_t)bb * NR_ + res) * 128;
#pragma unroll
    for (int i = 0; i < 8; ++i) {
      int c = l16 * 8 + i;
      s2[at * 128 + c] = (bf16)((xv[i] - mean) * rs * g2[c] + b2[c]);
    }
  }
  __syncthreads();

  int nn = lane & 15, mlo = (lane >> 4) << 3;

  // phase 2: s2 @ [lin1 | lin2] -> 16x512 (swizzled bf16 weights, K=128)
  for (int tn = w; tn < 32; tn += 8) {
    int nb0 = tn * 16;
    const bf16* WS = (nb0 < 256) ? lin1_s : lin2_s;
    int col = (nb0 < 256) ? nb0 : nb0 - 256;
    const bf16* wtile = WS + ((size_t)(col >> 4) * 4) * 512;
    v8f acc = {};
#pragma unroll
    for (int ks2 = 0; ks2 < 128; ks2 += 32) {
      v16bf a = load_A(s2 + ks2, 128);
      v16bf bfr = load_B_sw(wtile + (ks2 >> 5) * 512);
      acc = wmma_bf16(a, bfr, acc);
    }
#pragma unroll
    for (int r2 = 0; r2 < 8; ++r2) hraw[(r2 + mlo) * 512 + nb0 + nn] = acc[r2];
  }
  __syncthreads();

  // hb = silu(a1) * a2
  for (int i = t; i < 4096; i += 256) {
    int m = i >> 8, c = i & 255;
    float a1 = hraw[m * 512 + c], a2 = hraw[m * 512 + 256 + c];
    hbb[m * 256 + c] = (bf16)(a1 * sigf(a1) * a2);
  }
  __syncthreads();

  // phase 3: hb @ linb -> 16x128 (swizzled bf16 weights, K=256), gate, residual
  {
    const bf16* wtile = linb_s + (size_t)w * 8 * 512;
    v8f acc = {};
#pragma unroll
    for (int ks2 = 0; ks2 < 256; ks2 += 32) {
      v16bf a = load_A(hbb + ks2, 256);
      v16bf bfr = load_B_sw(wtile + (ks2 >> 5) * 512);
      acc = wmma_bf16(a, bfr, acc);
    }
#pragma unroll
    for (int r2 = 0; r2 < 8; ++r2) {
      int m = r2 + mlo;
      size_t arow = m0 + m;
      int c = w * 16 + nn;
      int res = resl[m];
      float tg = sig_tg[((size_t)bb * NR_ + res) * 128 + c];
      float mf = maskp[arow] ? 1.f : 0.f;
      xo[arow * 128 + c] += tg * acc[r2] * mf;
    }
  }
}

// =====================================================================
extern "C" void kernel_launch(void* const* d_in, const int* in_sizes, int n_in,
                              void* d_out, int out_size, void* d_ws, size_t ws_size,
                              hipStream_t stream) {
  (void)in_sizes; (void)n_in; (void)out_size; (void)ws_size;
  const float* atom_features   = (const float*)d_in[0];
  const float* atompos         = (const float*)d_in[1];
  const float* res_features    = (const float*)d_in[2];
  const float* rigids          = (const float*)d_in[3];
  const float* adaln_ln_cond_w = (const float*)d_in[5];
  const float* adaln_cond_W    = (const float*)d_in[6];
  const float* adaln_cond_b    = (const float*)d_in[7];
  const float* adaln_condnb_W  = (const float*)d_in[8];
  const float* Wq              = (const float*)d_in[9];
  const float* bq              = (const float*)d_in[10];
  const float* Wkv             = (const float*)d_in[11];
  const float* bij_ln_w        = (const float*)d_in[12];
  const float* bij_ln_b        = (const float*)d_in[13];
  const float* bij_W           = (const float*)d_in[14];
  const float* sgate_W         = (const float*)d_in[15];
  const float* sgate_b         = (const float*)d_in[16];
  const float* dvec_W          = (const float*)d_in[17];
  const float* dist_W          = (const float*)d_in[18];
  const float* tr_ln_cond_w    = (const float*)d_in[19];
  const float* tr_cond_W       = (const float*)d_in[20];
  const float* tr_cond_b       = (const float*)d_in[21];
  const float* tr_condnb_W     = (const float*)d_in[22];
  const float* tr_lin1_W       = (const float*)d_in[23];
  const float* tr_lin2_W       = (const float*)d_in[24];
  const float* tr_gate_W       = (const float*)d_in[25];
  const float* tr_gate_b       = (const float*)d_in[26];
  const float* tr_linb_W       = (const float*)d_in[27];
  const int*   atom_res_idx    = (const int*)d_in[28];
  const unsigned char* atompos_mask = (const unsigned char*)d_in[29];

  // workspace layout
  const size_t RESF = (size_t)B_ * NR_ * 128;          // 262144 floats
  const size_t AF   = (size_t)B_ * NA_ * 128;          // 4194304 elems
  float* sig_gate = (float*)d_ws;
  float* bias_w   = sig_gate + RESF;
  float* sig_g2   = bias_w + RESF;
  float* b2_w     = sig_g2 + RESF;
  float* sig_sg   = b2_w + RESF;
  float* sig_tg   = sig_sg + RESF;
  int*   nval     = (int*)(sig_tg + RESF);
  bf16*  af_w     = (bf16*)((char*)nval + 256);
  bf16*  q_w      = af_w + AF;
  bf16*  k_w      = q_w + AF;
  bf16*  v_t      = k_w + AF;        // transposed: [b][ch][atom]
  bf16*  wq_s     = v_t + AF;        // 16384
  bf16*  wkv_s    = wq_s + 16384;    // 32768
  bf16*  l1_s     = wkv_s + 32768;   // 32768
  bf16*  l2_s     = l1_s + 32768;    // 32768
  bf16*  lb_s     = l2_s + 32768;    // 32768

  float* xo  = (float*)d_out;                          // B*NA*128
  float* apf = xo + AF;                                // B*nb*32*128*16

  k_wswizzle<<<16384 / 256, 256, 0, stream>>>(Wq, 128, 128, wq_s);
  k_wswizzle<<<32768 / 256, 256, 0, stream>>>(Wkv, 128, 256, wkv_s);
  k_wswizzle<<<32768 / 256, 256, 0, stream>>>(tr_lin1_W, 128, 256, l1_s);
  k_wswizzle<<<32768 / 256, 256, 0, stream>>>(tr_lin2_W, 128, 256, l2_s);
  k_wswizzle<<<32768 / 256, 256, 0, stream>>>(tr_linb_W, 256, 128, lb_s);

  k_res_prep<<<B_ * NR_, 256, 0, stream>>>(
      res_features, adaln_ln_cond_w, adaln_cond_W, adaln_cond_b, adaln_condnb_W,
      tr_ln_cond_w, tr_cond_W, tr_cond_b, tr_condnb_W,
      sgate_W, sgate_b, tr_gate_W, tr_gate_b,
      sig_gate, bias_w, sig_g2, b2_w, sig_sg, sig_tg);

  k_nvalid<<<B_, 256, 0, stream>>>(atompos_mask, nval);

  k_atom_adaln<<<(B_ * NA_) / 8, 256, 0, stream>>>(
      atom_features, atom_res_idx, sig_gate, bias_w, af_w);

  k_qkv<<<dim3((B_ * NA_) / 16, 6), 128, 0, stream>>>(
      af_w, wq_s, bq, wkv_s, q_w, k_w, v_t);

  k_attn<<<dim3(NB_, B_), 256, 0, stream>>>(
      atom_features, atompos, rigids, atom_res_idx, atompos_mask, nval,
      dvec_W, dist_W, bij_ln_w, bij_ln_b, bij_W,
      q_w, k_w, v_t, sig_sg, xo, apf);

  k_trans<<<(B_ * NA_) / 16, 256, 0, stream>>>(
      l1_s, l2_s, lb_s, sig_g2, b2_w, sig_tg,
      atom_res_idx, atompos_mask, xo);
}
